// Attention_50809463111810
// MI455X (gfx1250) — matrix operand order repaired and checked by reading, not compiled, not run on hardware
//
#include <hip/hip_runtime.h>
#include <hip/hip_bf16.h>

// ---------------------------------------------------------------------------
// Attention forward for MI455X (gfx1250, wave32, WMMA).
//   B=4, S=1024, D=4096, H=32, HD=128
// bf16 WMMA (v_wmma_f32_16x16x32_bf16) for all matmuls, fp32 accumulate,
// fp32 softmax/RoPE. Intermediates held in bf16 workspace (memory roof
// ~26us @ 23.3 TB/s). All staging is register-batched and software-
// pipelined so global loads issue as clauses and overlap the WMMA phase
// instead of serializing on s_wait_loadcnt 0.
// ---------------------------------------------------------------------------

typedef __attribute__((ext_vector_type(16))) __bf16 v16bf;
typedef __attribute__((ext_vector_type(8)))  float  v8f;

constexpr int Bb = 4;
constexpr int Ss = 1024;
constexpr int Dd = 4096;
constexpr int Hh = 32;
constexpr int HD = 128;
constexpr size_t QKV_ELEMS = (size_t)Bb * Hh * Ss * HD;   // 16,777,216

static __device__ __forceinline__ v8f wmma_bf16(const v16bf& a, const v16bf& b, const v8f& c) {
    return __builtin_amdgcn_wmma_f32_16x16x32_bf16(false, a, false, b, (short)0, c, false, false);
}

// ---------------------------------------------------------------------------
// Kernel 1: fused QKV projection + RoPE.
//   C[m,n] = sum_k x[m,k] * w[n,k]   (x: (B*S, D) f32, w: (D, D) f32)
// grid = (D/128, B*S/128, 3), block = 256 (8 waves).
// 128x128 C tile per block; wave (wr,wc) owns 32x64 (2x4 WMMA tiles).
// K staged 64 at a time, f32->bf16, double-staged through registers.
// Epilogue: RoPE (z<2) via __shfl_xor pair exchange, bf16 store (B,H,S,HD).
// ---------------------------------------------------------------------------
__global__ __launch_bounds__(256)
void qkv_rope_kernel(const float* __restrict__ x,
                     const float* __restrict__ wq,
                     const float* __restrict__ wk,
                     const float* __restrict__ wv,
                     const float* __restrict__ fcos,
                     const float* __restrict__ fsin,
                     __bf16* __restrict__ qo,
                     __bf16* __restrict__ ko,
                     __bf16* __restrict__ vo)
{
    constexpr int KC = 64;
    __shared__ __attribute__((aligned(16))) __bf16 sA[128 * KC];
    __shared__ __attribute__((aligned(16))) __bf16 sB[128 * KC];

    const int nt = blockIdx.x, mt = blockIdx.y, z = blockIdx.z;
    const float* w    = (z == 0) ? wq : (z == 1) ? wk : wv;
    __bf16*      outp = (z == 0) ? qo : (z == 1) ? ko : vo;

    const int tid  = threadIdx.x;
    const int lane = tid & 31;
    const int wid  = tid >> 5;
    const int wr   = wid >> 1;     // 0..3 -> row offset wr*32
    const int wc   = wid & 1;      // 0..1 -> col offset wc*64
    const int m0   = mt * 128, n0 = nt * 128;
    const int r    = lane & 15, hh = lane >> 4;

    // staging geometry: 128 rows x 64 k as 2048 float4 chunks; each thread
    // owns column c4 of rows row0+16*i (i=0..7)
    const int row0 = tid >> 4;          // 0..15
    const int c4   = (tid & 15) * 4;    // 0..60

    v8f acc[2][4];
#pragma unroll
    for (int i = 0; i < 2; ++i)
#pragma unroll
        for (int j = 0; j < 4; ++j) acc[i][j] = (v8f){};

    // prologue: load first K-chunk into registers
    float4 xr[8], wreg[8];
#pragma unroll
    for (int i = 0; i < 8; ++i) {
        xr[i]   = *(const float4*)&x[(size_t)(m0 + row0 + 16 * i) * Dd + c4];
        wreg[i] = *(const float4*)&w[(size_t)(n0 + row0 + 16 * i) * Dd + c4];
    }

    for (int k0 = 0; k0 < Dd; k0 += KC) {
        // store register-staged chunk to LDS (f32 -> bf16)
#pragma unroll
        for (int i = 0; i < 8; ++i) {
            __bf16* pa = &sA[(row0 + 16 * i) * KC + c4];
            pa[0] = (__bf16)xr[i].x; pa[1] = (__bf16)xr[i].y;
            pa[2] = (__bf16)xr[i].z; pa[3] = (__bf16)xr[i].w;
            __bf16* pb = &sB[(row0 + 16 * i) * KC + c4];
            pb[0] = (__bf16)wreg[i].x; pb[1] = (__bf16)wreg[i].y;
            pb[2] = (__bf16)wreg[i].z; pb[3] = (__bf16)wreg[i].w;
        }
        __syncthreads();

        // pipeline: issue next chunk's global loads now, consumed next iter
        if (k0 + KC < Dd) {
#pragma unroll
            for (int i = 0; i < 8; ++i) {
                xr[i]   = *(const float4*)&x[(size_t)(m0 + row0 + 16 * i) * Dd + k0 + KC + c4];
                wreg[i] = *(const float4*)&w[(size_t)(n0 + row0 + 16 * i) * Dd + k0 + KC + c4];
            }
            if (k0 + 2 * KC < Dd) {  // L2 warm hint (global_prefetch_b8)
                __builtin_prefetch(&x[(size_t)(m0 + row0) * Dd + k0 + 2 * KC + c4], 0, 0);
                __builtin_prefetch(&w[(size_t)(n0 + row0) * Dd + k0 + 2 * KC + c4], 0, 0);
            }
        }

        // compute: 2 k-subchunks of 32, 8 WMMAs each
#pragma unroll
        for (int kc = 0; kc < 2; ++kc) {
            v16bf af[2];
#pragma unroll
            for (int rt = 0; rt < 2; ++rt) {
                const __bf16* ap = &sA[(wr * 32 + rt * 16 + r) * KC + kc * 32];
#pragma unroll
                for (int i = 0; i < 8; ++i) {
                    af[rt][i]     = ap[hh * 8 + i];
                    af[rt][8 + i] = ap[16 + hh * 8 + i];
                }
            }
            v16bf bfr[4];
#pragma unroll
            for (int ct = 0; ct < 4; ++ct) {
                const __bf16* bp = &sB[(wc * 64 + ct * 16 + r) * KC + kc * 32];
#pragma unroll
                for (int i = 0; i < 16; ++i) bfr[ct][i] = bp[hh * 8 + i + (i >= 8 ? 8 : 0)];
            }
#pragma unroll
            for (int rt = 0; rt < 2; ++rt)
#pragma unroll
                for (int ct = 0; ct < 4; ++ct)
                    acc[rt][ct] = wmma_bf16(af[rt], bfr[ct], acc[rt][ct]);
        }
        __syncthreads();
    }

    // Epilogue: RoPE (q,k) + bf16 store into (B,H,S,HD)
#pragma unroll
    for (int rt = 0; rt < 2; ++rt) {
#pragma unroll
        for (int ct = 0; ct < 4; ++ct) {
            int n    = n0 + wc * 64 + ct * 16 + r;
            int head = n >> 7;
            int hd   = n & 127;
#pragma unroll
            for (int j = 0; j < 8; ++j) {
                float val = acc[rt][ct][j];
                float pv  = __shfl_xor(val, 1, 32);  // rotation partner column
                int m    = m0 + wr * 32 + rt * 16 + hh * 8 + j;
                int bidx = m >> 10;
                int s    = m & 1023;
                float outv = val;
                if (z < 2) {
                    int   fi = hd >> 1;
                    float c  = fcos[s * 64 + fi];
                    float sn = fsin[s * 64 + fi];
                    outv = (hd & 1) ? (pv * sn + val * c)   // imag: re*s + im*c
                                    : (val * c - pv * sn);  // real: re*c - im*s
                }
                outp[(((size_t)bidx * Hh + head) * Ss + s) * HD + hd] = (__bf16)outv;
            }
        }
    }
}

// ---------------------------------------------------------------------------
// Kernel 2: causal flash attention over bf16 q/k/v (layout (B,H,S,HD)).
// grid = (S/128, B*H), block = 256. Each wave owns 16 query rows with a
// 16x128 fp32 accumulator; 64x128 K/V tiles staged in LDS, register-
// pipelined across key tiles. Online softmax via half-wave shfl reductions.
// ---------------------------------------------------------------------------
__global__ __launch_bounds__(256)
void attn_kernel(const __bf16* __restrict__ q,
                 const __bf16* __restrict__ k,
                 const __bf16* __restrict__ v,
                 __bf16* __restrict__ o)
{
    __shared__ __attribute__((aligned(16))) __bf16 sK[64 * 128];
    __shared__ __attribute__((aligned(16))) __bf16 sV[64 * 128];
    __shared__ __attribute__((aligned(16))) __bf16 sP[8][16 * 64];

    const int qt   = blockIdx.x;
    const int bh   = blockIdx.y;
    const int bidx = bh >> 5;
    const int head = bh & 31;
    const int tid  = threadIdx.x;
    const int lane = tid & 31;
    const int wid  = tid >> 5;
    const int r    = lane & 15, hh = lane >> 4;
    const size_t base = (size_t)bh * Ss * HD;
    const int qrow0 = qt * 128 + wid * 16;

    // staging geometry: 64 rows x 128 hd as 1024 8-bf16 chunks
    const int srow0 = tid >> 4;          // 0..15 (+16*i)
    const int sc8   = (tid & 15) * 8;    // 0..120

    // Preload this wave's Q A-fragments (16 rows x 128 hd)
    v16bf qa[4];
#pragma unroll
    for (int kc = 0; kc < 4; ++kc) {
        const __bf16* qp = &q[base + (size_t)(qrow0 + r) * HD + kc * 32];
#pragma unroll
        for (int i = 0; i < 8; ++i) {
            qa[kc][i]     = qp[hh * 8 + i];
            qa[kc][8 + i] = qp[16 + hh * 8 + i];
        }
    }

    v8f oacc[8];
#pragma unroll
    for (int i = 0; i < 8; ++i) oacc[i] = (v8f){};
    float mrow[8], lrow[8];
#pragma unroll
    for (int j = 0; j < 8; ++j) { mrow[j] = -__builtin_inff(); lrow[j] = 0.0f; }

    const float scale  = 0.08838834764831845f;  // 1/sqrt(128)
    const int   ktiles = qt * 2 + 2;            // causal bound

    // prologue: first K/V tile into registers
    uint4 kr[4], vr[4];
#pragma unroll
    for (int i = 0; i < 4; ++i) {
        const size_t g = base + (size_t)(srow0 + 16 * i) * HD + sc8;
        kr[i] = *(const uint4*)&k[g];
        vr[i] = *(const uint4*)&v[g];
    }

    for (int kt = 0; kt < ktiles; ++kt) {
        // store register-staged tile
#pragma unroll
        for (int i = 0; i < 4; ++i) {
            *(uint4*)&sK[(srow0 + 16 * i) * 128 + sc8] = kr[i];
            *(uint4*)&sV[(srow0 + 16 * i) * 128 + sc8] = vr[i];
        }
        __syncthreads();

        // pipeline: next tile's loads issue before compute
        if (kt + 1 < ktiles) {
#pragma unroll
            for (int i = 0; i < 4; ++i) {
                const size_t g = base + (size_t)((kt + 1) * 64 + srow0 + 16 * i) * HD + sc8;
                kr[i] = *(const uint4*)&k[g];
                vr[i] = *(const uint4*)&v[g];
            }
        }

        // scores: Q(16x128) @ K_tile^T -> 4 subtiles of 16 keys
        v8f sacc[4];
#pragma unroll
        for (int ct = 0; ct < 4; ++ct) {
            v8f a = (v8f){};
#pragma unroll
            for (int kc = 0; kc < 4; ++kc) {
                v16bf bfr;
                const __bf16* kp = &sK[(ct * 16 + r) * 128 + kc * 32];
#pragma unroll
                for (int i = 0; i < 16; ++i) bfr[i] = kp[hh * 8 + i + (i >= 8 ? 8 : 0)];
                a = wmma_bf16(qa[kc], bfr, a);
            }
            sacc[ct] = a;
        }

        // scale + causal mask + row max
        float pm[8];
#pragma unroll
        for (int j = 0; j < 8; ++j) pm[j] = -__builtin_inff();
#pragma unroll
        for (int ct = 0; ct < 4; ++ct) {
            int key = kt * 64 + ct * 16 + r;
#pragma unroll
            for (int j = 0; j < 8; ++j) {
                int   qs = qrow0 + hh * 8 + j;
                float sv = sacc[ct][j] * scale;
                if (key > qs) sv = -__builtin_inff();
                sacc[ct][j] = sv;
                pm[j] = fmaxf(pm[j], sv);
            }
        }
#pragma unroll
        for (int msk = 1; msk < 16; msk <<= 1)
#pragma unroll
            for (int j = 0; j < 8; ++j)
                pm[j] = fmaxf(pm[j], __shfl_xor(pm[j], msk, 32));

        float alpha[8];
#pragma unroll
        for (int j = 0; j < 8; ++j) {
            float mn = fmaxf(mrow[j], pm[j]);
            alpha[j] = __expf(mrow[j] - mn);
            mrow[j]  = mn;
        }
        float ps[8];
#pragma unroll
        for (int j = 0; j < 8; ++j) ps[j] = 0.0f;
#pragma unroll
        for (int ct = 0; ct < 4; ++ct)
#pragma unroll
            for (int j = 0; j < 8; ++j) {
                float p = __expf(sacc[ct][j] - mrow[j]);
                sacc[ct][j] = p;
                ps[j] += p;
            }
#pragma unroll
        for (int msk = 1; msk < 16; msk <<= 1)
#pragma unroll
            for (int j = 0; j < 8; ++j) ps[j] += __shfl_xor(ps[j], msk, 32);
#pragma unroll
        for (int j = 0; j < 8; ++j) lrow[j] = lrow[j] * alpha[j] + ps[j];

        // rescale running output
#pragma unroll
        for (int ct8 = 0; ct8 < 8; ++ct8)
#pragma unroll
            for (int j = 0; j < 8; ++j) oacc[ct8][j] *= alpha[j];

        // C-layout -> A-layout relayout of P via wave-private LDS
        // (same-wave LDS ops are in-order; no barrier needed)
#pragma unroll
        for (int ct = 0; ct < 4; ++ct)
#pragma unroll
            for (int j = 0; j < 8; ++j)
                sP[wid][(hh * 8 + j) * 64 + ct * 16 + r] = (__bf16)sacc[ct][j];

        // O += P(16x64) @ V(64x128)
#pragma unroll
        for (int kc2 = 0; kc2 < 2; ++kc2) {
            v16bf pa;
            const __bf16* pp = &sP[wid][r * 64 + kc2 * 32];
#pragma unroll
            for (int i = 0; i < 8; ++i) {
                pa[i]     = pp[hh * 8 + i];
                pa[8 + i] = pp[16 + hh * 8 + i];
            }
#pragma unroll
            for (int ct8 = 0; ct8 < 8; ++ct8) {
                v16bf bfr;
                const __bf16* vp = &sV[(kc2 * 32) * 128 + ct8 * 16 + r];
#pragma unroll
                for (int i = 0; i < 16; ++i) bfr[i] = vp[(hh * 8 + i + (i >= 8 ? 8 : 0)) * 128];
                oacc[ct8] = wmma_bf16(pa, bfr, oacc[ct8]);
            }
        }
        __syncthreads();
    }

    // normalize and store bf16 into (B,S,H,HD) == (B*S, D)
#pragma unroll
    for (int ct8 = 0; ct8 < 8; ++ct8) {
        int hd = ct8 * 16 + r;
#pragma unroll
        for (int j = 0; j < 8; ++j) {
            int qs = qrow0 + hh * 8 + j;
            float ov = oacc[ct8][j] / lrow[j];
            o[((size_t)bidx * Ss + qs) * Dd + head * 128 + hd] = (__bf16)ov;
        }
    }
}

// ---------------------------------------------------------------------------
// Kernel 3: output projection  out[m,n] = sum_k attn[m,k] * wo[n,k]  (f32 out)
// ---------------------------------------------------------------------------
__global__ __launch_bounds__(256)
void out_proj_kernel(const __bf16* __restrict__ a,
                     const float* __restrict__ wo,
                     float* __restrict__ out)
{
    constexpr int KC = 64;
    __shared__ __attribute__((aligned(16))) __bf16 sA[128 * KC];
    __shared__ __attribute__((aligned(16))) __bf16 sB[128 * KC];

    const int nt = blockIdx.x, mt = blockIdx.y;
    const int tid  = threadIdx.x;
    const int lane = tid & 31;
    const int wid  = tid >> 5;
    const int wr   = wid >> 1;
    const int wc   = wid & 1;
    const int m0   = mt * 128, n0 = nt * 128;
    const int r    = lane & 15, hh = lane >> 4;

    // staging geometry
    const int arow0 = tid >> 3;          // A: 1024 8-bf16 chunks (+32*i)
    const int ac8   = (tid & 7) * 8;
    const int wrow0 = tid >> 4;          // W: 2048 float4 chunks (+16*i)
    const int wc4   = (tid & 15) * 4;

    v8f acc[2][4];
#pragma unroll
    for (int i = 0; i < 2; ++i)
#pragma unroll
        for (int j = 0; j < 4; ++j) acc[i][j] = (v8f){};

    uint4 ar[4];
    float4 wreg[8];
#pragma unroll
    for (int i = 0; i < 4; ++i)
        ar[i] = *(const uint4*)&a[(size_t)(m0 + arow0 + 32 * i) * Dd + ac8];
#pragma unroll
    for (int i = 0; i < 8; ++i)
        wreg[i] = *(const float4*)&wo[(size_t)(n0 + wrow0 + 16 * i) * Dd + wc4];

    for (int k0 = 0; k0 < Dd; k0 += KC) {
#pragma unroll
        for (int i = 0; i < 4; ++i)
            *(uint4*)&sA[(arow0 + 32 * i) * KC + ac8] = ar[i];
#pragma unroll
        for (int i = 0; i < 8; ++i) {
            __bf16* pb = &sB[(wrow0 + 16 * i) * KC + wc4];
            pb[0] = (__bf16)wreg[i].x; pb[1] = (__bf16)wreg[i].y;
            pb[2] = (__bf16)wreg[i].z; pb[3] = (__bf16)wreg[i].w;
        }
        __syncthreads();

        if (k0 + KC < Dd) {
#pragma unroll
            for (int i = 0; i < 4; ++i)
                ar[i] = *(const uint4*)&a[(size_t)(m0 + arow0 + 32 * i) * Dd + k0 + KC + ac8];
#pragma unroll
            for (int i = 0; i < 8; ++i)
                wreg[i] = *(const float4*)&wo[(size_t)(n0 + wrow0 + 16 * i) * Dd + k0 + KC + wc4];
        }

#pragma unroll
        for (int kc = 0; kc < 2; ++kc) {
            v16bf af[2];
#pragma unroll
            for (int rt = 0; rt < 2; ++rt) {
                const __bf16* ap = &sA[(wr * 32 + rt * 16 + r) * KC + kc * 32];
#pragma unroll
                for (int i = 0; i < 8; ++i) {
                    af[rt][i]     = ap[hh * 8 + i];
                    af[rt][8 + i] = ap[16 + hh * 8 + i];
                }
            }
            v16bf bfr[4];
#pragma unroll
            for (int ct = 0; ct < 4; ++ct) {
                const __bf16* bp = &sB[(wc * 64 + ct * 16 + r) * KC + kc * 32];
#pragma unroll
                for (int i = 0; i < 16; ++i) bfr[ct][i] = bp[hh * 8 + i + (i >= 8 ? 8 : 0)];
            }
#pragma unroll
            for (int rt = 0; rt < 2; ++rt)
#pragma unroll
                for (int ct = 0; ct < 4; ++ct)
                    acc[rt][ct] = wmma_bf16(af[rt], bfr[ct], acc[rt][ct]);
        }
        __syncthreads();
    }

#pragma unroll
    for (int rt = 0; rt < 2; ++rt)
#pragma unroll
        for (int ct = 0; ct < 4; ++ct) {
            int n = n0 + wc * 64 + ct * 16 + r;
#pragma unroll
            for (int j = 0; j < 8; ++j) {
                int m = m0 + wr * 32 + rt * 16 + hh * 8 + j;
                out[(size_t)m * Dd + n] = acc[rt][ct][j];
            }
        }
}

// ---------------------------------------------------------------------------
extern "C" void kernel_launch(void* const* d_in, const int* in_sizes, int n_in,
                              void* d_out, int out_size, void* d_ws, size_t ws_size,
                              hipStream_t stream) {
    const float* x    = (const float*)d_in[0];
    const float* wq   = (const float*)d_in[1];
    const float* wk   = (const float*)d_in[2];
    const float* wv   = (const float*)d_in[3];
    const float* wo   = (const float*)d_in[4];
    const float* fcos = (const float*)d_in[5];
    const float* fsin = (const float*)d_in[6];
    // d_in[7] (mask) unused: causal mask applied analytically.

    __bf16* qws = (__bf16*)d_ws;
    __bf16* kws = qws + QKV_ELEMS;
    __bf16* vws = kws + QKV_ELEMS;
    __bf16* aws = vws + QKV_ELEMS;

    dim3 blk(256);
    qkv_rope_kernel<<<dim3(Dd / 128, (Bb * Ss) / 128, 3), blk, 0, stream>>>(
        x, wq, wk, wv, fcos, fsin, qws, kws, vws);
    attn_kernel<<<dim3(Ss / 128, Bb * Hh), blk, 0, stream>>>(qws, kws, vws, aws);
    out_proj_kernel<<<dim3(Dd / 128, (Bb * Ss) / 128), blk, 0, stream>>>(
        aws, wo, (float*)d_out);
}
